// MoLE_59957743452295
// MI455X (gfx1250) — compile-verified, matrix operand order checked by reading, not run on hardware
//
#include <hip/hip_runtime.h>

typedef __attribute__((ext_vector_type(16))) __bf16 v16bf;
typedef __attribute__((ext_vector_type(8)))  __bf16 v8bf;
typedef __attribute__((ext_vector_type(8)))  float  v8f;
typedef __attribute__((ext_vector_type(4)))  float  v4f;
typedef __attribute__((ext_vector_type(4)))  unsigned int u32x4;
typedef __attribute__((ext_vector_type(8)))  int i32x8;
typedef __attribute__((ext_vector_type(4)))  int i32x4;

#define D_IN   2048
#define D_OUT  2048
#define NTOK   8192
#define NEXP   8
#define RANKL  8
#define KLORA  64
#define KTOT   2112          // 2048 + 64 (lora K-extension), row stride of split arrays
#define SCALING 1.0f         // alpha/rank = 8/8

// ---------- f32 -> (hi,lo) bf16 split, bit-level only ----------
__device__ __forceinline__ void split_f32u(float f, unsigned short& h, unsigned short& l) {
  unsigned u = __builtin_bit_cast(unsigned, f);
  h = (unsigned short)(u >> 16);                         // truncated hi
  float hf = __builtin_bit_cast(float, u & 0xFFFF0000u);
  float r  = f - hf;                                     // residual
  unsigned ur = __builtin_bit_cast(unsigned, r);
  l = (unsigned short)((ur + 0x7FFFu + ((ur >> 16) & 1u)) >> 16);  // RNE
}

__device__ __forceinline__ v16bf cat8(v8bf a, v8bf b) {
  v16bf r;
#pragma unroll
  for (int i = 0; i < 8; ++i) { r[i] = a[i]; r[i + 8] = b[i]; }
  return r;
}

// =====================================================================
// K0/K1: split a row-major [nrows x 2048] f32 matrix into hi/lo bf16
// arrays with row stride KTOT (tail columns left for the K-extension).
// =====================================================================
__global__ __launch_bounds__(256) void split_matrix_kernel(
    const float* __restrict__ src,
    unsigned short* __restrict__ Dh, unsigned short* __restrict__ Dl)
{
  size_t base = ((size_t)blockIdx.x * 256 + threadIdx.x) * 8;
  size_t row = base >> 11;            // /2048
  size_t k   = base & 2047;
  const float* s = src + base;
  unsigned short* ph = Dh + row * KTOT + k;
  unsigned short* pl = Dl + row * KTOT + k;
#pragma unroll
  for (int i = 0; i < 2; ++i) {
    v4f v = *(const v4f*)(s + i * 4);
    ushort4 hs, ls;
    split_f32u(v[0], hs.x, ls.x);
    split_f32u(v[1], hs.y, ls.y);
    split_f32u(v[2], hs.z, ls.z);
    split_f32u(v[3], hs.w, ls.w);
    *(ushort4*)(ph + i * 4) = hs;
    *(ushort4*)(pl + i * 4) = ls;
  }
}

// =====================================================================
// K2: Bcat[(e,r), o] = Bm[e][o][r] -> tail columns 2048..2111 of Wh/Wl
// =====================================================================
__global__ __launch_bounds__(256) void split_bm_kernel(
    const float* __restrict__ Bm,
    unsigned short* __restrict__ Wh, unsigned short* __restrict__ Wl)
{
  int t = blockIdx.x * 256 + threadIdx.x;   // 0 .. 2048*64-1
  int o  = t >> 6;
  int kk = t & 63;
  int e = kk >> 3, r = kk & 7;
  float v = Bm[((size_t)e * D_OUT + o) * RANKL + r];
  unsigned short h, l; split_f32u(v, h, l);
  Wh[(size_t)o * KTOT + D_IN + kk] = h;
  Wl[(size_t)o * KTOT + D_IN + kk] = l;
}

// =====================================================================
// K3: router logits + softmax + h = x@A^T.
//   wout[t][e] (f32, d_out tail);  g -> hi/lo bf16 into Xh/Xl tail cols.
// =====================================================================
__global__ __launch_bounds__(256) void mole_router_kernel(
    const float* __restrict__ x, const float* __restrict__ Wr,
    const float* __restrict__ br, const float* __restrict__ A,
    unsigned short* __restrict__ Xh, unsigned short* __restrict__ Xl,
    float* __restrict__ wout)
{
  __shared__ float Ms[72 * 128];
  __shared__ float sh_h[8][64];
  __shared__ float sh_w[8][8];

  const int tid  = threadIdx.x;
  const int lane = tid & 31;
  const int wave = tid >> 5;
  const int token = blockIdx.x * 8 + wave;

  float acc[72];
#pragma unroll
  for (int j = 0; j < 72; ++j) acc[j] = 0.0f;

  for (int d0 = 0; d0 < D_IN; d0 += 128) {
    __syncthreads();
#pragma unroll
    for (int i = 0; i < 36; ++i) {
      int idx = tid + i * 256;
      int j  = idx >> 7;
      int dd = idx & 127;
      Ms[idx] = (j < 8) ? Wr[j * D_IN + d0 + dd]
                        : A[(j - 8) * D_IN + d0 + dd];
    }
    __syncthreads();
#pragma unroll
    for (int it = 0; it < 2; ++it) {
      int db = it * 64 + lane * 2;
      float2 xv = *(const float2*)(x + (size_t)token * D_IN + d0 + db);
#pragma unroll
      for (int j = 0; j < 72; ++j) {
        float2 m = *(const float2*)(Ms + j * 128 + db);
        acc[j] = fmaf(xv.y, m.y, fmaf(xv.x, m.x, acc[j]));
      }
    }
  }

#pragma unroll
  for (int j = 0; j < 72; ++j) {
    float v = acc[j];
    v += __shfl_xor(v, 16, 32);
    v += __shfl_xor(v,  8, 32);
    v += __shfl_xor(v,  4, 32);
    v += __shfl_xor(v,  2, 32);
    v += __shfl_xor(v,  1, 32);
    acc[j] = v;
  }

  if (lane == 0) {
    float lg[8], we[8];
    float mx = -3.0e38f;
#pragma unroll
    for (int e = 0; e < 8; ++e) { lg[e] = acc[e] + br[e]; mx = fmaxf(mx, lg[e]); }
    float s = 0.0f;
#pragma unroll
    for (int e = 0; e < 8; ++e) { we[e] = expf(lg[e] - mx); s += we[e]; }
    float inv = 1.0f / s;
#pragma unroll
    for (int e = 0; e < 8; ++e) {
      float wn = we[e] * inv;
      wout[(size_t)token * NEXP + e] = wn;
      sh_w[wave][e] = wn;
    }
#pragma unroll
    for (int j = 0; j < 64; ++j) sh_h[wave][j] = acc[8 + j];
  }
  __syncthreads();
  {
    int j0 = lane, j1 = lane + 32;
    float g0 = SCALING * sh_w[wave][j0 >> 3] * sh_h[wave][j0];
    float g1 = SCALING * sh_w[wave][j1 >> 3] * sh_h[wave][j1];
    unsigned short h, l;
    split_f32u(g0, h, l);
    Xh[(size_t)token * KTOT + D_IN + j0] = h;
    Xl[(size_t)token * KTOT + D_IN + j0] = l;
    split_f32u(g1, h, l);
    Xh[(size_t)token * KTOT + D_IN + j1] = h;
    Xl[(size_t)token * KTOT + D_IN + j1] = l;
  }
}

// =====================================================================
// TDM: load one [128 rows x 32 halves] tile (row stride KTOT halves)
// into LDS at ldsOff, padded to 80B row pitch (pad 4 dwords every 16).
// D# per CDNA5 ISA ch.8: group0 = ctl/lds/global addr, group1 = dims.
// =====================================================================
__device__ __forceinline__ void tdm_load_tile(const unsigned short* gsrc,
                                              unsigned ldsOff) {
  unsigned long long ga = (unsigned long long)gsrc;
  u32x4 g0;
  g0[0] = 1u;                                   // count=1 (valid), user mode
  g0[1] = ldsOff;                               // lds_addr (bytes)
  g0[2] = (unsigned)ga;                         // global_addr[31:0]
  g0[3] = (unsigned)(ga >> 32) | 0x80000000u;   // global_addr[56:32] | type=2
  i32x8 g1;
  g1[0] = (1 << 16)        // data_size = 2 bytes
        | (1 << 20)        // pad_enable
        | (3 << 22)        // pad_interval: every 16 dwords (64B = one row)
        | (3 << 25);       // pad_amount: 4 dwords (16B) -> 80B pitch
  g1[1] = (int)(2112u << 16);       // tensor_dim0 = 2112 (bits 79:48, low part)
  g1[2] = (int)(8192u << 16);       // tensor_dim0 hi=0 | tensor_dim1 = 8192
  g1[3] = (int)(32u << 16);         // tensor_dim1 hi=0 | tile_dim0 = 32
  g1[4] = 128;                      // tile_dim1 = 128, tile_dim2 = 0
  g1[5] = 2112;                     // tensor_dim0_stride = 2112 (48b, low)
  g1[6] = 0;                        // stride0 hi | stride1 low
  g1[7] = 0;
  i32x4 z4 = {0, 0, 0, 0};
#if defined(__clang_major__) && (__clang_major__ >= 23)
  i32x8 z8 = {0, 0, 0, 0, 0, 0, 0, 0};
  __builtin_amdgcn_tensor_load_to_lds(g0, g1, z4, z4, z8, 0);
#else
  __builtin_amdgcn_tensor_load_to_lds(g0, g1, z4, z4, 0);
#endif
}

// =====================================================================
// K4: out = [x|g]_split @ [Wf^T|Bcat]_split + bias, bf16x3 WMMA.
// M=8192, N=2048, K=2112 (66 chunks of 32).
// Block tile 128x128, 8 waves (2Mx4N), wave tile 64x32 -> 24 WMMA/chunk.
// TDM double-buffered: wave0 DMAs chunk k+1 while all waves WMMA chunk k.
// One barrier per chunk.
// =====================================================================
#define BM 128
#define BN 128
#define KC 32
#define KS 40   // LDS row stride in halves (80B) = TDM pad output pitch

__global__ __launch_bounds__(256) void mole_gemm_kernel(
    const unsigned short* __restrict__ Xh, const unsigned short* __restrict__ Xl,
    const unsigned short* __restrict__ Wh, const unsigned short* __restrict__ Wl,
    const float* __restrict__ bias, float* __restrict__ out)
{
  // [buf][mat][tile]: mat 0=Xh 1=Xl 2=Wh 3=Wl ; 80 KB total
  __shared__ __align__(16) unsigned short SB[2][4][BM * KS];

  const int tid   = threadIdx.x;
  const int lane  = tid & 31;
  const int wave  = tid >> 5;
  const int waveM = wave >> 2;     // 0..1 : rows waveM*64 .. +64
  const int waveN = wave & 3;      // 0..3 : cols waveN*32 .. +32
  const int l16   = lane & 15;
  const int hiH   = lane >> 4;

  const int rowBase = blockIdx.y * BM;
  const int colBase = blockIdx.x * BN;

  const size_t xOff = (size_t)rowBase * KTOT;
  const size_t wOff = (size_t)colBase * KTOT;

  v8f acc[4][2];
#pragma unroll
  for (int mi = 0; mi < 4; ++mi)
#pragma unroll
    for (int ni = 0; ni < 2; ++ni)
#pragma unroll
      for (int i = 0; i < 8; ++i) acc[mi][ni][i] = 0.0f;

  unsigned ldsOff[2][4];
#pragma unroll
  for (int b = 0; b < 2; ++b)
#pragma unroll
    for (int m = 0; m < 4; ++m)
      ldsOff[b][m] = (unsigned)(unsigned long long)(void*)&SB[b][m][0];

  // prime the pipeline: chunk 0 -> buffer 0
  if (wave == 0) {
    tdm_load_tile(Xh + xOff, ldsOff[0][0]);
    tdm_load_tile(Xl + xOff, ldsOff[0][1]);
    tdm_load_tile(Wh + wOff, ldsOff[0][2]);
    tdm_load_tile(Wl + wOff, ldsOff[0][3]);
  }

  for (int kc = 0; kc < 66; ++kc) {
    const int buf = kc & 1;
    if (wave == 0) __builtin_amdgcn_s_wait_tensorcnt(0);  // chunk kc landed
    __syncthreads();   // publish tile; also: all waves done reading buf^1

    if (wave == 0 && kc < 65) {
      const size_t k1 = (size_t)(kc + 1) * KC;
      tdm_load_tile(Xh + xOff + k1, ldsOff[buf ^ 1][0]);
      tdm_load_tile(Xl + xOff + k1, ldsOff[buf ^ 1][1]);
      tdm_load_tile(Wh + wOff + k1, ldsOff[buf ^ 1][2]);
      tdm_load_tile(Wl + wOff + k1, ldsOff[buf ^ 1][3]);
    }

    const unsigned short* XhS = SB[buf][0];
    const unsigned short* XlS = SB[buf][1];
    const unsigned short* WhS = SB[buf][2];
    const unsigned short* WlS = SB[buf][3];

    // ---- fetch WMMA operands (dense 16-bit ISA layout) ----
    v16bf ah[4], al[4], bh[2], bl[2];
#pragma unroll
    for (int mi = 0; mi < 4; ++mi) {
      int row  = waveM * 64 + mi * 16 + l16;
      int base = row * KS + hiH * 8;
      ah[mi] = cat8(*(const v8bf*)(XhS + base), *(const v8bf*)(XhS + base + 16));
      al[mi] = cat8(*(const v8bf*)(XlS + base), *(const v8bf*)(XlS + base + 16));
    }
#pragma unroll
    for (int ni = 0; ni < 2; ++ni) {
      int col  = waveN * 32 + ni * 16 + l16;
      int base = col * KS + hiH * 8;
      bh[ni] = cat8(*(const v8bf*)(WhS + base), *(const v8bf*)(WhS + base + 16));
      bl[ni] = cat8(*(const v8bf*)(WlS + base), *(const v8bf*)(WlS + base + 16));
    }

    // ---- bf16x3: hi*hi + hi*lo + lo*hi, f32 accumulate ----
#pragma unroll
    for (int mi = 0; mi < 4; ++mi)
#pragma unroll
      for (int ni = 0; ni < 2; ++ni) {
        acc[mi][ni] = __builtin_amdgcn_wmma_f32_16x16x32_bf16(
            false, ah[mi], false, bh[ni], (short)0, acc[mi][ni], false, false);
        acc[mi][ni] = __builtin_amdgcn_wmma_f32_16x16x32_bf16(
            false, ah[mi], false, bl[ni], (short)0, acc[mi][ni], false, false);
        acc[mi][ni] = __builtin_amdgcn_wmma_f32_16x16x32_bf16(
            false, al[mi], false, bh[ni], (short)0, acc[mi][ni], false, false);
      }
  }

  // ---- epilogue: C layout VGPR i -> M = i + 8*hiH, N = l16 ----
#pragma unroll
  for (int ni = 0; ni < 2; ++ni) {
    int col = colBase + waveN * 32 + ni * 16 + l16;
    float b = bias[col];
#pragma unroll
    for (int mi = 0; mi < 4; ++mi) {
      int row0 = rowBase + waveM * 64 + mi * 16 + hiH * 8;
#pragma unroll
      for (int i = 0; i < 8; ++i) {
        out[(size_t)(row0 + i) * D_OUT + col] = acc[mi][ni][i] + b;
      }
    }
  }
}

extern "C" void kernel_launch(void* const* d_in, const int* in_sizes, int n_in,
                              void* d_out, int out_size, void* d_ws, size_t ws_size,
                              hipStream_t stream) {
  (void)in_sizes; (void)n_in; (void)out_size; (void)ws_size;
  const float* x   = (const float*)d_in[0];
  const float* Wf  = (const float*)d_in[1];
  const float* bfv = (const float*)d_in[2];
  const float* Wr  = (const float*)d_in[3];
  const float* br  = (const float*)d_in[4];
  const float* A   = (const float*)d_in[5];
  const float* Bm  = (const float*)d_in[6];

  float* out  = (float*)d_out;                       // [8192, 2048]
  float* wout = out + (size_t)NTOK * D_OUT;          // [8192, 8]

  // workspace layout (halves): Xh, Xl [8192*2112]; Wh, Wl [2048*2112]  (~86.5 MB)
  const size_t XE = (size_t)NTOK * KTOT;
  const size_t WE = (size_t)D_OUT * KTOT;
  unsigned short* Xh = (unsigned short*)d_ws;
  unsigned short* Xl = Xh + XE;
  unsigned short* Wh = Xl + XE;
  unsigned short* Wl = Wh + WE;

  split_matrix_kernel<<<(NTOK * D_IN) / (256 * 8), 256, 0, stream>>>(x, Xh, Xl);
  split_matrix_kernel<<<(D_OUT * D_IN) / (256 * 8), 256, 0, stream>>>(Wf, Wh, Wl);
  split_bm_kernel<<<(D_OUT * KLORA) / 256, 256, 0, stream>>>(Bm, Wh, Wl);
  mole_router_kernel<<<NTOK / 8, 256, 0, stream>>>(x, Wr, br, A, Xh, Xl, wout);
  mole_gemm_kernel<<<dim3(D_OUT / BN, NTOK / BM), 256, 0, stream>>>(
      Xh, Xl, Wh, Wl, bfv, out);
}
